// MultiHeadGatedCrossAttention_72103910965775
// MI455X (gfx1250) — matrix operand order repaired and checked by reading, not compiled, not run on hardware
//
#include <hip/hip_runtime.h>
#include <hip/hip_bf16.h>
#include <stdint.h>

// ---- problem constants (match reference) ----
#define DMODEL 512
#define NHEAD 8
#define DHEAD 64
#define BB 2
#define LQ 2048
#define LK 4096

typedef __attribute__((ext_vector_type(16))) _Float16 v16h;
typedef __attribute__((ext_vector_type(8)))  float    v8f;

// ---------------------------------------------------------------------------
// Fragment loaders (wave32, 16-bit WMMA operand layouts per CDNA5 ISA 7.12.2)
// A-matrix 16x32 (MxK): lane holds row M=lane%16; u32 e2 -> K pair:
//   K = (e2>=4?16:0) + half*8 + (e2&3)*2   => u32 off = ((e2>>2)<<3)+(half<<2)+(e2&3)
//   (two groups of 4 contiguous dwords -> 2x ds_load_b128)
// B-matrix 32x16 (KxN): lane holds col N=lane%16; u32 v -> K = (v>=4?16:0) + half*8 + (v&3)*2
//   => u32 off = ((v>>2)<<3)+(half<<2)+(v&3)   (two groups of 4 contiguous dwords -> 2x ds_load_b128)
// ---------------------------------------------------------------------------
__device__ __forceinline__ v16h frag_ld_a(const uint32_t* rowBaseU32, int half) {
  union { v16h h; uint32_t u[8]; } f;
#pragma unroll
  for (int e2 = 0; e2 < 8; ++e2)
    f.u[e2] = rowBaseU32[((e2 >> 2) << 3) + (half << 2) + (e2 & 3)];
  return f.h;
}

__device__ __forceinline__ v16h frag_ld_b(const uint32_t* colBaseU32, int half) {
  union { v16h h; uint32_t u[8]; } f;
#pragma unroll
  for (int v = 0; v < 8; ++v)
    f.u[v] = colBaseU32[((v >> 2) << 3) + (half << 2) + (v & 3)];
  return f.h;
}

// ---------------------------------------------------------------------------
// Stage 0: f32 -> f16 conversion (inputs + weights)
// ---------------------------------------------------------------------------
__global__ void f32_to_f16_kernel(const float* __restrict__ src,
                                  _Float16* __restrict__ dst, int n) {
  for (int i = blockIdx.x * blockDim.x + threadIdx.x; i < n;
       i += gridDim.x * blockDim.x)
    dst[i] = (_Float16)src[i];
}

// gbias[b*LK + k] = mask ? log(clip(gate,1e-6)) : -1e30
__global__ void gbias_kernel(const float* __restrict__ gate,
                             const int* __restrict__ mask,
                             float* __restrict__ gb, int n) {
  int i = blockIdx.x * blockDim.x + threadIdx.x;
  if (i < n)
    gb[i] = (mask[i] == 0) ? -1.0e30f : __logf(fmaxf(gate[i], 1e-6f));
}

// ---------------------------------------------------------------------------
// Stage 1/3: GEMM  Y[M,512] = X[M,512] @ W[512,512]^T + bias
//   128 threads = 4 waves; block tile 128(M) x 64(N); K staged in LDS by 32.
//   Each wave owns 32 M-rows: 2 A-frags x 4 B-frags = 8 WMMAs per K-step
//   (1.5 ds_load_b128 per WMMA, A and B register reuse x2).
//   headMode=1: write f16 to outH in [B, H, Lseq, 64] layout (for attention)
//   headMode=0: write f32 to outF row-major [M,512] (final output)
// ---------------------------------------------------------------------------
__global__ __launch_bounds__(128) void proj_gemm_kernel(
    const _Float16* __restrict__ X, const _Float16* __restrict__ W,
    const float* __restrict__ bias, _Float16* __restrict__ outH,
    float* __restrict__ outF, int M, int Lseq, int headMode) {
  __shared__ __align__(16) _Float16 sA[128 * 32];
  __shared__ __align__(16) _Float16 sW[64 * 32];
  const int t = threadIdx.x;
  const int w = t >> 5, lane = t & 31, half = lane >> 4, rl = lane & 15;
  const int m0 = blockIdx.x * 128;
  const int n0 = blockIdx.y * 64;

  v8f acc[2][4] = {};

  for (int kk = 0; kk < DMODEL; kk += 32) {
#pragma unroll
    for (int rep = 0; rep < 4; ++rep) {        // 512 uint4 of A
      int i = t + rep * 128;
      int row = i >> 2, c8 = i & 3;
      ((uint4*)sA)[i] =
          ((const uint4*)(X + (size_t)(m0 + row) * DMODEL + kk))[c8];
    }
#pragma unroll
    for (int rep = 0; rep < 2; ++rep) {        // 256 uint4 of W slice
      int i = t + rep * 128;
      int row = i >> 2, c8 = i & 3;
      ((uint4*)sW)[i] =
          ((const uint4*)(W + (size_t)(n0 + row) * DMODEL + kk))[c8];
    }
    __syncthreads();

    v16h a0 = frag_ld_a((const uint32_t*)sA + (w * 32 + rl) * 16, half);
    v16h a1 = frag_ld_a((const uint32_t*)sA + (w * 32 + 16 + rl) * 16, half);
#pragma unroll
    for (int nt = 0; nt < 4; ++nt) {
      v16h bf = frag_ld_b((const uint32_t*)sW + (nt * 16 + rl) * 16, half);
      acc[0][nt] = __builtin_amdgcn_wmma_f32_16x16x32_f16(
          false, a0, false, bf, (short)0, acc[0][nt], false, false);
      acc[1][nt] = __builtin_amdgcn_wmma_f32_16x16x32_f16(
          false, a1, false, bf, (short)0, acc[1][nt], false, false);
    }
    __syncthreads();
  }

#pragma unroll
  for (int mt = 0; mt < 2; ++mt)
#pragma unroll
    for (int nt = 0; nt < 4; ++nt) {
      int n = n0 + nt * 16 + rl;
      float bv = bias[n];
#pragma unroll
      for (int v = 0; v < 8; ++v) {
        float val = acc[mt][nt][v] + bv;
        int m = m0 + w * 32 + mt * 16 + v + half * 8;
        if (headMode) {
          int bidx = m / Lseq;
          int l = m - bidx * Lseq;
          int h = n >> 6, d = n & 63;
          outH[(((size_t)bidx * NHEAD + h) * Lseq + l) * DHEAD + d] =
              (_Float16)val;
        } else {
          outF[(size_t)m * DMODEL + n] = val;
        }
      }
    }
}

// ---------------------------------------------------------------------------
// Stage 2: flash attention per (b, h).  grid = (LQ/64, H, B), 128 thr = 4 waves.
// Each wave owns 16 q rows.  Stream K/V in 32-key tiles through LDS; V staged
// transposed so P@V B-fragments are contiguous.  Online softmax in f32.
// ---------------------------------------------------------------------------
__global__ __launch_bounds__(128) void attn_kernel(
    const _Float16* __restrict__ Qp, const _Float16* __restrict__ Kp,
    const _Float16* __restrict__ Vp, const float* __restrict__ gbias,
    _Float16* __restrict__ attnOut) {
  __shared__ __align__(16) _Float16 sK[32 * 64];   // [kvrow][d]
  __shared__ __align__(16) _Float16 sVt[64 * 32];  // [d][kvrow]
  __shared__ __align__(16) _Float16 sP[4][16 * 32];

  const int t = threadIdx.x;
  const int w = t >> 5, lane = t & 31, half = lane >> 4, rl = lane & 15;
  const int h = blockIdx.y, b = blockIdx.z;
  const int q0 = blockIdx.x * 64 + w * 16;
  const size_t bh = (size_t)b * NHEAD + h;

  // Q fragments for this wave's 16 rows (d 0..31 and 32..63), loaded once.
  const uint32_t* qU =
      (const uint32_t*)(Qp + (bh * LQ + q0 + rl) * DHEAD);
  v16h qa0 = frag_ld_a(qU, half);
  v16h qa1 = frag_ld_a(qU + 16, half);

  v8f o[4] = {};
  float mrun[8], lrun[8];
#pragma unroll
  for (int v = 0; v < 8; ++v) { mrun[v] = -1.0e30f; lrun[v] = 0.0f; }

  for (int j0 = 0; j0 < LK; j0 += 32) {
    const _Float16* kg = Kp + (bh * LK + j0) * DHEAD;
    const _Float16* vg = Vp + (bh * LK + j0) * DHEAD;
#pragma unroll
    for (int rep = 0; rep < 2; ++rep) {   // 256 uint4 each of K and V
      int i = t + rep * 128;
      ((uint4*)sK)[i] = ((const uint4*)kg)[i];
      union { uint4 u4; _Float16 hh[8]; } tv;
      tv.u4 = ((const uint4*)vg)[i];
      int row = i >> 3, c8 = i & 7;
#pragma unroll
      for (int e = 0; e < 8; ++e)
        sVt[(c8 * 8 + e) * 32 + row] = tv.hh[e];   // transpose into LDS
    }
    __syncthreads();

    // scores: S = (Q K^T) * 0.125 + gbias   (16 x 32 tile per wave)
    float sc[2][8];
#pragma unroll
    for (int nt = 0; nt < 2; ++nt) {
      const uint32_t* kU = (const uint32_t*)sK + (nt * 16 + rl) * 32;
      v16h kb0 = frag_ld_b(kU, half);
      v16h kb1 = frag_ld_b(kU + 16, half);
      v8f s = {};
      s = __builtin_amdgcn_wmma_f32_16x16x32_f16(false, qa0, false, kb0,
                                                 (short)0, s, false, false);
      s = __builtin_amdgcn_wmma_f32_16x16x32_f16(false, qa1, false, kb1,
                                                 (short)0, s, false, false);
      float gb = gbias[(size_t)b * LK + j0 + nt * 16 + rl];
#pragma unroll
      for (int v = 0; v < 8; ++v) sc[nt][v] = s[v] * 0.125f + gb;
    }

    // online softmax: rows live in (v, half); cols across 16-lane groups
    float alpha[8];
#pragma unroll
    for (int v = 0; v < 8; ++v) {
      float rm = fmaxf(sc[0][v], sc[1][v]);
      rm = fmaxf(rm, __shfl_xor(rm, 1));
      rm = fmaxf(rm, __shfl_xor(rm, 2));
      rm = fmaxf(rm, __shfl_xor(rm, 4));
      rm = fmaxf(rm, __shfl_xor(rm, 8));
      float nm = fmaxf(mrun[v], rm);
      alpha[v] = __expf(mrun[v] - nm);
      mrun[v] = nm;
      float p0 = (sc[0][v] <= -1.0e29f) ? 0.0f : __expf(sc[0][v] - nm);
      float p1 = (sc[1][v] <= -1.0e29f) ? 0.0f : __expf(sc[1][v] - nm);
      sc[0][v] = p0; sc[1][v] = p1;
      float rs = p0 + p1;
      rs += __shfl_xor(rs, 1);
      rs += __shfl_xor(rs, 2);
      rs += __shfl_xor(rs, 4);
      rs += __shfl_xor(rs, 8);
      lrun[v] = lrun[v] * alpha[v] + rs;
    }
#pragma unroll
    for (int dt = 0; dt < 4; ++dt)
#pragma unroll
      for (int v = 0; v < 8; ++v) o[dt][v] *= alpha[v];

    // C-layout -> A-layout for P via per-wave LDS buffer (wave-local fence)
#pragma unroll
    for (int nt = 0; nt < 2; ++nt)
#pragma unroll
      for (int v = 0; v < 8; ++v)
        sP[w][(v + half * 8) * 32 + nt * 16 + rl] = (_Float16)sc[nt][v];
    asm volatile("s_wait_dscnt 0" ::: "memory");

    v16h pa = frag_ld_a((const uint32_t*)&sP[w][0] + rl * 16, half);
#pragma unroll
    for (int dt = 0; dt < 4; ++dt) {
      v16h vb = frag_ld_b((const uint32_t*)sVt + (dt * 16 + rl) * 16, half);
      o[dt] = __builtin_amdgcn_wmma_f32_16x16x32_f16(
          false, pa, false, vb, (short)0, o[dt], false, false);
    }
    __syncthreads();   // protect sK/sVt for next tile's staging
  }

  // finalize: divide by l, write f16 in [B, LQ, H*64] for the output GEMM
#pragma unroll
  for (int dt = 0; dt < 4; ++dt)
#pragma unroll
    for (int v = 0; v < 8; ++v) {
      float l = lrun[v];
      float val = (l > 0.0f) ? o[dt][v] / l : 0.0f;
      int qrow = q0 + v + half * 8;
      int d = h * DHEAD + dt * 16 + rl;
      attnOut[((size_t)b * LQ + qrow) * DMODEL + d] = (_Float16)val;
    }
}

// ---------------------------------------------------------------------------
extern "C" void kernel_launch(void* const* d_in, const int* in_sizes, int n_in,
                              void* d_out, int out_size, void* d_ws,
                              size_t ws_size, hipStream_t stream) {
  const float* q       = (const float*)d_in[0];
  const float* kv      = (const float*)d_in[1];
  const float* gate    = (const float*)d_in[2];
  const int*   mask    = (const int*)d_in[3];
  const float* Wq      = (const float*)d_in[4];
  const float* bq      = (const float*)d_in[5];
  const float* Wk      = (const float*)d_in[6];
  const float* bk      = (const float*)d_in[7];
  const float* Wv      = (const float*)d_in[8];
  const float* bv      = (const float*)d_in[9];
  const float* Wo      = (const float*)d_in[10];
  const float* bo      = (const float*)d_in[11];
  float* out = (float*)d_out;

  const int nQ  = BB * LQ * DMODEL;   // 2,097,152
  const int nKV = BB * LK * DMODEL;   // 4,194,304
  const int nW  = DMODEL * DMODEL;    // 262,144
  const int nG  = BB * LK;            // 8,192

  char* p = (char*)d_ws;
  auto carve = [&](size_t bytes) -> void* {
    void* r = (void*)p;
    p += (bytes + 255) & ~(size_t)255;
    return r;
  };
  _Float16* q16    = (_Float16*)carve((size_t)nQ * 2);
  _Float16* kv16   = (_Float16*)carve((size_t)nKV * 2);
  _Float16* Wq16   = (_Float16*)carve((size_t)nW * 2);
  _Float16* Wk16   = (_Float16*)carve((size_t)nW * 2);
  _Float16* Wv16   = (_Float16*)carve((size_t)nW * 2);
  _Float16* Wo16   = (_Float16*)carve((size_t)nW * 2);
  _Float16* Qp     = (_Float16*)carve((size_t)nQ * 2);   // [B,H,LQ,64]
  _Float16* Kp     = (_Float16*)carve((size_t)nKV * 2);  // [B,H,LK,64]
  _Float16* Vp     = (_Float16*)carve((size_t)nKV * 2);  // [B,H,LK,64]
  _Float16* attn16 = (_Float16*)carve((size_t)nQ * 2);   // [B,LQ,512]
  float*    gb     = (float*)carve((size_t)nG * 4);

  // Stage 0: conversions + gate/mask fusion
  f32_to_f16_kernel<<<1024, 256, 0, stream>>>(q,  q16,  nQ);
  f32_to_f16_kernel<<<1024, 256, 0, stream>>>(kv, kv16, nKV);
  f32_to_f16_kernel<<<256,  256, 0, stream>>>(Wq, Wq16, nW);
  f32_to_f16_kernel<<<256,  256, 0, stream>>>(Wk, Wk16, nW);
  f32_to_f16_kernel<<<256,  256, 0, stream>>>(Wv, Wv16, nW);
  f32_to_f16_kernel<<<256,  256, 0, stream>>>(Wo, Wo16, nW);
  gbias_kernel<<<(nG + 255) / 256, 256, 0, stream>>>(gate, mask, gb, nG);

  // Stage 1: projections (head layout)
  proj_gemm_kernel<<<dim3(BB * LQ / 128, DMODEL / 64), 128, 0, stream>>>(
      q16, Wq16, bq, Qp, nullptr, BB * LQ, LQ, 1);
  proj_gemm_kernel<<<dim3(BB * LK / 128, DMODEL / 64), 128, 0, stream>>>(
      kv16, Wk16, bk, Kp, nullptr, BB * LK, LK, 1);
  proj_gemm_kernel<<<dim3(BB * LK / 128, DMODEL / 64), 128, 0, stream>>>(
      kv16, Wv16, bv, Vp, nullptr, BB * LK, LK, 1);

  // Stage 2: flash attention
  attn_kernel<<<dim3(LQ / 64, NHEAD, BB), 128, 0, stream>>>(Qp, Kp, Vp, gb,
                                                            attn16);

  // Stage 3: output projection -> f32 d_out
  proj_gemm_kernel<<<dim3(BB * LQ / 128, DMODEL / 64), 128, 0, stream>>>(
      attn16, Wo16, bo, nullptr, out, BB * LQ, LQ, 0);
}
